// Attention_79766132621772
// MI455X (gfx1250) — compile-verified, matrix operand order checked by reading, not run on hardware
//
#include <hip/hip_runtime.h>

// Problem constants (from reference): B=2, L=2048, D=2048, H=16, HD=128
#define BATCH  2
#define SEQ    2048
#define DMODEL 2048
#define NHEADS 16
#define HEADD  128

typedef _Float16 h8v  __attribute__((ext_vector_type(8)));
typedef _Float16 v16h __attribute__((ext_vector_type(16)));
typedef float    v8f  __attribute__((ext_vector_type(8)));
typedef unsigned u32x4 __attribute__((ext_vector_type(4)));
typedef unsigned u32x8 __attribute__((ext_vector_type(8)));

union AF { v16h v; h8v h8[2]; _Float16 h[16]; };
union CF { v8f  v; float f[8]; };

// ln(10000)/128 for RoPE inverse frequencies: 10000^(-d'/128) = exp(-c*d')
#define ROPE_C 0.0719557843283949f

// Load one 16-bit A/B WMMA fragment from LDS.
// CDNA5 layout (16-bit, 16x32): lanes 0-15 hold K = [0..7] then [16..23],
// lanes 16-31 hold K = [8..15] then [24..31].  Caller passes base+row*stride+kb
// where kb = (lane&16)?8:0.  Both halves are 8 contiguous halfs = ds_load_b128.
__device__ __forceinline__ v16h lds_frag16(const _Float16* p) {
  AF a;
  a.h8[0] = *(const h8v*)(p);
  a.h8[1] = *(const h8v*)(p + 16);
  return a.v;
}

#define WMMA_F16(A, Bm, C) \
  __builtin_amdgcn_wmma_f32_16x16x32_f16(false, (A), false, (Bm), (short)0, (C), false, false)

// Async global->LDS copy of 16 bytes per lane (gfx1250, tracked by ASYNCcnt).
// lds_off: addrspace(3) byte offset (low 32 bits of a generic LDS pointer).
__device__ __forceinline__ void async_copy_b128(unsigned lds_off, const void* gptr) {
  asm volatile("global_load_async_to_lds_b128 %0, %1, off"
               :: "v"(lds_off), "v"(gptr) : "memory");
}
__device__ __forceinline__ void async_wait0() {
  asm volatile("s_wait_asynccnt 0" ::: "memory");
}

// Tensor Data Mover: DMA a 2D tile (16-bit elements) global -> LDS.
// Rows are tile_d0 elements; LDS rows padded +16B: rows of 64 DWORDs
// (pad_interval code 5) followed by 4 DWORDs pad (pad_amount code 3),
// matching the [row][HEADD+8] half layouts used below (256B + 16B pad).
// Issued by one wave only (TDM ignores EXEC); tracked by TENSORcnt.
__device__ __forceinline__ void tdm_load_2d(unsigned lds_off, const void* gaddr,
                                            unsigned tensor_d0, unsigned tensor_d1,
                                            unsigned tile_d0, unsigned tile_d1,
                                            unsigned d0_stride) {
  const unsigned long long ga = (unsigned long long)(uintptr_t)gaddr;
  u32x4 g0;
  g0[0] = 1u;                                   // count=1 (valid user D#)
  g0[1] = lds_off;                              // lds_addr (bytes)
  g0[2] = (unsigned)ga;                         // global_addr[31:0]
  g0[3] = (unsigned)(ga >> 32) | (2u << 30);    // global_addr[56:32] | type=2
  u32x8 g1;
  g1[0] = (1u << 16) | (1u << 20) | (5u << 22) | (3u << 25);
          // data_size=2B | pad_enable | pad_interval=64dw | pad_amount=4dw
  g1[1] = (tensor_d0 & 0xFFFFu) << 16;                          // dim0[15:0]
  g1[2] = (tensor_d0 >> 16) | ((tensor_d1 & 0xFFFFu) << 16);    // dim0 hi, dim1 lo
  g1[3] = (tensor_d1 >> 16) | ((tile_d0 & 0xFFFFu) << 16);      // dim1 hi, tile0
  g1[4] = tile_d1 & 0xFFFFu;                                    // tile1 (tile2=0)
  g1[5] = d0_stride;                                            // dim0 stride lo
  g1[6] = 0u;
  g1[7] = 0u;
  asm volatile("tensor_load_to_lds %0, %1" :: "s"(g0), "s"(g1) : "memory");
}

// ---------------------------------------------------------------------------
// Kernel 1: qkv = x @ w_qkv   (M=4096, K=2048, N=6144), fused RoPE on q,k.
// Output scattered as f16 into q/k/v buffers of shape [B,H,L,HD].
// (f32->f16 conversion forces the VALU staging path here.)
// ---------------------------------------------------------------------------
__global__ __launch_bounds__(256)
void qkv_rope_kernel(const float* __restrict__ x, const float* __restrict__ w,
                     _Float16* __restrict__ qh, _Float16* __restrict__ kh,
                     _Float16* __restrict__ vh) {
  const int K = DMODEL;
  const int N = 3 * DMODEL;
  const int BM = 128, BN = 64, BK = 32;

  __shared__ __align__(16) _Float16 As[128][BK + 8];  // row-major, K contiguous
  __shared__ __align__(16) _Float16 Bs[64][BK + 8];   // transposed: [n][k]

  const int tid  = threadIdx.x;
  const int lane = tid & 31;
  const int wid  = tid >> 5;
  const int wm   = wid & 3;              // wave grid 4 (M) x 2 (N)
  const int wn   = wid >> 2;
  const int m0   = blockIdx.y * BM;
  const int n0   = blockIdx.x * BN;
  const int kb    = (lane & 16) ? 8 : 0;
  const int rbase = (lane & 16) ? 8 : 0;
  const int lc    = lane & 15;

  CF c[2][2];
  for (int i = 0; i < 2; i++)
    for (int j = 0; j < 2; j++)
      for (int r = 0; r < 8; r++) c[i][j].f[r] = 0.0f;

  for (int kt = 0; kt < K; kt += BK) {
    __syncthreads();
    for (int idx = tid; idx < BM * BK; idx += 256) {
      int r = idx >> 5, cc = idx & 31;
      As[r][cc] = (_Float16)x[(size_t)(m0 + r) * K + kt + cc];
    }
    for (int idx = tid; idx < BK * BN; idx += 256) {
      int nn = idx & 63, kk = idx >> 6;
      Bs[nn][kk] = (_Float16)w[(size_t)(kt + kk) * N + n0 + nn];
    }
    __syncthreads();

    v16h a0 = lds_frag16(&As[wm * 32 + lc][kb]);
    v16h a1 = lds_frag16(&As[wm * 32 + 16 + lc][kb]);
    v16h b0 = lds_frag16(&Bs[wn * 32 + lc][kb]);
    v16h b1 = lds_frag16(&Bs[wn * 32 + 16 + lc][kb]);
    c[0][0].v = WMMA_F16(a0, b0, c[0][0].v);
    c[0][1].v = WMMA_F16(a0, b1, c[0][1].v);
    c[1][0].v = WMMA_F16(a1, b0, c[1][0].v);
    c[1][1].v = WMMA_F16(a1, b1, c[1][1].v);
  }

  // Epilogue: RoPE on q,k sections; scatter f16 into [B,H,L,HD].
  for (int mi = 0; mi < 2; mi++) {
    for (int ni = 0; ni < 2; ni++) {
      const int col = n0 + wn * 32 + ni * 16 + lc;       // [0, 6144)
      const int s   = col >> 11;                         // 0=q 1=k 2=v
      const int rem = col & 2047;
      const int h   = rem >> 7;
      const int d   = rem & 127;
      // 10000^(-(d&~1)/128) via a single v_exp_f32 (not __powf)
      const float inv = __expf(-ROPE_C * (float)(d & ~1));
      for (int r = 0; r < 8; r++) {
        float val = c[mi][ni].f[r];
        // adjacent head-dim lives in the adjacent lane (C layout: N = lane&15)
        float partner = __shfl_xor(val, 1, 32);
        const int row = m0 + wm * 32 + mi * 16 + rbase + r;
        const int bb  = row >> 11;
        const int l   = row & 2047;
        float outv = val;
        if (s < 2) {
          float ang = (float)l * inv;
          float sn, cs;
          __sincosf(ang, &sn, &cs);
          outv = (d & 1) ? fmaf(val, cs, partner * sn)
                         : fmaf(val, cs, -partner * sn);
        }
        _Float16* dst = (s == 0) ? qh : (s == 1) ? kh : vh;
        dst[(((size_t)bb * NHEADS + h) * SEQ + l) * HEADD + d] = (_Float16)outv;
      }
    }
  }
}

// ---------------------------------------------------------------------------
// Kernel 2: flash attention per (b, h).  Workgroup = 128 query rows
// (8 waves x 16 rows), streaming 32-key chunks with online softmax.
// Q tile and K chunks are DMA'd by the Tensor Data Mover (TENSORcnt);
// V is staged transposed through VGPRs (TDM cannot transpose).
// Per chunk per wave: 8 WMMAs (S = Q K^T) + 8 WMMAs (O += P V).
// ---------------------------------------------------------------------------
__global__ __launch_bounds__(256)
void flash_attn_kernel(const _Float16* __restrict__ qh,
                       const _Float16* __restrict__ kh,
                       const _Float16* __restrict__ vh,
                       _Float16* __restrict__ oh) {
  const int qblk = blockIdx.x;
  const int h    = blockIdx.y;
  const int b    = blockIdx.z;
  const int tid  = threadIdx.x;
  const int lane = tid & 31;
  const int wid  = tid >> 5;
  const int lc    = lane & 15;
  const int kb    = (lane & 16) ? 8 : 0;
  const int rbase = (lane & 16) ? 8 : 0;
  const float scale = 0.08838834764831845f;   // 1/sqrt(128)

  __shared__ __align__(16) _Float16 Qs[128][HEADD + 8];   // [q_row][d], 272B rows
  __shared__ __align__(16) _Float16 Ks[32][HEADD + 8];    // [key][d]
  __shared__ __align__(16) _Float16 Vt[HEADD][32 + 8];    // [d][key]
  __shared__ __align__(16) _Float16 Ps[8][16][32 + 8];    // per-wave P tile

  const size_t bh = (size_t)b * NHEADS + h;
  const _Float16* qbase = qh + bh * SEQ * HEADD;
  const _Float16* kbase = kh + bh * SEQ * HEADD;
  const _Float16* vbase = vh + bh * SEQ * HEADD;
  const int q0 = qblk * 128;

  const unsigned ldsQ = (unsigned)(uintptr_t)&Qs[0][0];
  const unsigned ldsK = (unsigned)(uintptr_t)&Ks[0][0];

  // Whole Q tile (128 x 128 halfs) in one TDM descriptor, issued by wave 0.
  if (wid == 0)
    tdm_load_2d(ldsQ, qbase + (size_t)q0 * HEADD,
                HEADD, SEQ, HEADD, 128, HEADD);

  CF o[8];
  float m_r[8], l_r[8];
  for (int f = 0; f < 8; f++)
    for (int r = 0; r < 8; r++) o[f].f[r] = 0.0f;
  for (int r = 0; r < 8; r++) { m_r[r] = -1e30f; l_r[r] = 0.0f; }

  for (int kc = 0; kc < SEQ; kc += 32) {
    __syncthreads();   // previous chunk fully consumed by all waves
    // K chunk (32 x 128 halfs): one TDM descriptor per chunk.
    if (wid == 0)
      tdm_load_2d(ldsK, kbase + (size_t)kc * HEADD,
                  HEADD, SEQ, HEADD, 32, HEADD);
    // V chunk: staged transposed so PV B-fragments are contiguous in key.
    for (int idx = tid; idx < 32 * HEADD; idx += 256) {
      int r = idx >> 7, dd = idx & 127;
      Vt[dd][r] = vbase[(size_t)(kc + r) * HEADD + dd];
    }
    if (kc + 32 < SEQ)   // global_prefetch_b8 for the next V chunk
      __builtin_prefetch(vbase + (size_t)(kc + 32) * HEADD + tid * 16, 0, 1);
    if (wid == 0)
      __builtin_amdgcn_s_wait_tensorcnt(0);   // Q tile (1st iter) + K chunk
    __syncthreads();

    // S[16 x 32] = Q_tile(16 x 128) @ K_chunk^T(128 x 32)
    CF s0, s1;
    for (int r = 0; r < 8; r++) { s0.f[r] = 0.0f; s1.f[r] = 0.0f; }
    for (int d0 = 0; d0 < HEADD; d0 += 32) {
      v16h qa = lds_frag16(&Qs[wid * 16 + lc][d0 + kb]);
      v16h k0 = lds_frag16(&Ks[lc][d0 + kb]);
      v16h k1 = lds_frag16(&Ks[16 + lc][d0 + kb]);
      s0.v = WMMA_F16(qa, k0, s0.v);
      s1.v = WMMA_F16(qa, k1, s1.v);
    }

    // online softmax: row reductions across the 16 lanes holding each row
    float mc[8], alpha[8], rs[8];
    for (int r = 0; r < 8; r++) {
      s0.f[r] *= scale;
      s1.f[r] *= scale;
      mc[r] = fmaxf(s0.f[r], s1.f[r]);
    }
    for (int mask = 1; mask <= 8; mask <<= 1)
      for (int r = 0; r < 8; r++)
        mc[r] = fmaxf(mc[r], __shfl_xor(mc[r], mask, 32));
    for (int r = 0; r < 8; r++) {
      float mn = fmaxf(m_r[r], mc[r]);
      alpha[r] = __expf(m_r[r] - mn);
      float p0 = __expf(s0.f[r] - mn);
      float p1 = __expf(s1.f[r] - mn);
      s0.f[r] = p0; s1.f[r] = p1;
      rs[r] = p0 + p1;
      m_r[r] = mn;
    }
    for (int mask = 1; mask <= 8; mask <<= 1)
      for (int r = 0; r < 8; r++)
        rs[r] += __shfl_xor(rs[r], mask, 32);
    for (int r = 0; r < 8; r++) l_r[r] = l_r[r] * alpha[r] + rs[r];
    for (int f = 0; f < 8; f++)
      for (int r = 0; r < 8; r++) o[f].f[r] *= alpha[r];

    // C-layout -> A-layout transpose of P through wave-private LDS
    for (int r = 0; r < 8; r++) {
      Ps[wid][rbase + r][lc]      = (_Float16)s0.f[r];
      Ps[wid][rbase + r][16 + lc] = (_Float16)s1.f[r];
    }
    asm volatile("s_wait_dscnt 0" ::: "memory");   // wave-private: no barrier

    // O[16 x 128] += P(16 x 32) @ V_chunk(32 x 128)
    v16h pa = lds_frag16(&Ps[wid][lc][kb]);
    for (int f = 0; f < 8; f++) {
      v16h vb = lds_frag16(&Vt[f * 16 + lc][kb]);
      o[f].v = WMMA_F16(pa, vb, o[f].v);
    }
  }

  // normalize and write [B,L,D] f16 (heads re-interleaved at col h*HD+d)
  const size_t outbase = (size_t)b * SEQ * DMODEL + (size_t)h * HEADD;
  for (int f = 0; f < 8; f++) {
    const int dcol = f * 16 + lc;
    for (int r = 0; r < 8; r++) {
      const int qr = q0 + wid * 16 + rbase + r;
      float val = o[f].f[r] / l_r[r];
      oh[outbase + (size_t)qr * DMODEL + dcol] = (_Float16)val;
    }
  }
}

// ---------------------------------------------------------------------------
// Kernel 3: out = attn @ w_proj   (M=4096, K=2048, N=2048), f32 result.
// A is already f16 -> staged via async global->LDS DMA (ASYNCcnt);
// B needs f32->f16 conversion (VALU path).
// ---------------------------------------------------------------------------
__global__ __launch_bounds__(256)
void proj_kernel(const _Float16* __restrict__ a, const float* __restrict__ w,
                 float* __restrict__ out) {
  const int K = DMODEL, N = DMODEL;
  const int BM = 128, BN = 64, BK = 32;

  __shared__ __align__(16) _Float16 As[128][BK + 8];   // 80B rows
  __shared__ __align__(16) _Float16 Bs[64][BK + 8];

  const int tid  = threadIdx.x;
  const int lane = tid & 31;
  const int wid  = tid >> 5;
  const int wm   = wid & 3;
  const int wn   = wid >> 2;
  const int m0   = blockIdx.y * BM;
  const int n0   = blockIdx.x * BN;
  const int kb    = (lane & 16) ? 8 : 0;
  const int rbase = (lane & 16) ? 8 : 0;
  const int lc    = lane & 15;

  CF c[2][2];
  for (int i = 0; i < 2; i++)
    for (int j = 0; j < 2; j++)
      for (int r = 0; r < 8; r++) c[i][j].f[r] = 0.0f;

  const unsigned ldsA = (unsigned)(uintptr_t)&As[0][0];

  for (int kt = 0; kt < K; kt += BK) {
    __syncthreads();
    // A tile: async DMA, 128x32 halfs = 512 x 16B chunks.
    for (int it = 0; it < 2; ++it) {
      const int chunk = tid + it * 256;          // [0, 512)
      const int row = chunk >> 2;                // 4 x 16B per row
      const int c8  = chunk & 3;
      async_copy_b128(ldsA + row * (BK + 8) * 2 + c8 * 16,
                      a + (size_t)(m0 + row) * K + kt + c8 * 8);
    }
    for (int idx = tid; idx < BK * BN; idx += 256) {
      int nn = idx & 63, kk = idx >> 6;
      Bs[nn][kk] = (_Float16)w[(size_t)(kt + kk) * N + n0 + nn];
    }
    async_wait0();
    __syncthreads();

    v16h a0 = lds_frag16(&As[wm * 32 + lc][kb]);
    v16h a1 = lds_frag16(&As[wm * 32 + 16 + lc][kb]);
    v16h b0 = lds_frag16(&Bs[wn * 32 + lc][kb]);
    v16h b1 = lds_frag16(&Bs[wn * 32 + 16 + lc][kb]);
    c[0][0].v = WMMA_F16(a0, b0, c[0][0].v);
    c[0][1].v = WMMA_F16(a0, b1, c[0][1].v);
    c[1][0].v = WMMA_F16(a1, b0, c[1][0].v);
    c[1][1].v = WMMA_F16(a1, b1, c[1][1].v);
  }

  for (int mi = 0; mi < 2; mi++) {
    for (int ni = 0; ni < 2; ni++) {
      const int col = n0 + wn * 32 + ni * 16 + lc;
      for (int r = 0; r < 8; r++) {
        const int row = m0 + wm * 32 + mi * 16 + rbase + r;
        out[(size_t)row * N + col] = c[mi][ni].f[r];
      }
    }
  }
}

// ---------------------------------------------------------------------------
extern "C" void kernel_launch(void* const* d_in, const int* in_sizes, int n_in,
                              void* d_out, int out_size, void* d_ws, size_t ws_size,
                              hipStream_t stream) {
  const float* x      = (const float*)d_in[0];
  const float* w_qkv  = (const float*)d_in[1];
  const float* w_proj = (const float*)d_in[2];
  float* out = (float*)d_out;

  const size_t per = (size_t)BATCH * NHEADS * SEQ * HEADD;  // 8,388,608 elems
  _Float16* qh = (_Float16*)d_ws;
  _Float16* kh = qh + per;
  _Float16* vh = kh + per;
  _Float16* ah = vh + per;   // attention output, [B*L][D] f16

  dim3 blk(256);
  qkv_rope_kernel<<<dim3((3 * DMODEL) / 64, (BATCH * SEQ) / 128), blk, 0, stream>>>(
      x, w_qkv, qh, kh, vh);
  flash_attn_kernel<<<dim3(SEQ / 128, NHEADS, BATCH), blk, 0, stream>>>(
      qh, kh, vh, ah);
  proj_kernel<<<dim3(DMODEL / 64, (BATCH * SEQ) / 128), blk, 0, stream>>>(
      ah, w_proj, out);
}
